// GINEStyleGraphTransformer_266287972603
// MI455X (gfx1250) — matrix operand order repaired and checked by reading, not compiled, not run on hardware
//
#include <hip/hip_runtime.h>
#include <hip/hip_bf16.h>
#include <math.h>
#include <stdint.h>

// ---------------------------------------------------------------------------
// Problem dimensions (fixed by the reference)
// ---------------------------------------------------------------------------
#define N_NODES 16384
#define N_EDGES 262144
#define HID     128
#define HEADS   8
#define DH      16
#define NLAYERS 4
#define NGRAPHS 64

typedef __attribute__((ext_vector_type(16))) _Float16 v16h;
typedef __attribute__((ext_vector_type(8)))  _Float16 v8h;
typedef __attribute__((ext_vector_type(8)))  float    v8f;
typedef __attribute__((ext_vector_type(4)))  int      v4i;

struct HalfPair { v8h lo; v8h hi; };
union Frag16 { v16h v; HalfPair p; };

__device__ inline v8h zero8h() {
    v8h z;
    #pragma unroll
    for (int j = 0; j < 8; ++j) z[j] = (_Float16)0.0f;
    return z;
}

// ---------------------------------------------------------------------------
// CDNA5 async global->LDS copy (guarded so the file always compiles).
// Builtin signature (from probe diagnostics): pointer params are to 4xi32
// vectors; (global src, LDS dst, imm offset, imm cpol).
// ---------------------------------------------------------------------------
#if __has_builtin(__builtin_amdgcn_global_load_async_to_lds_b128) && \
    __has_builtin(__builtin_amdgcn_s_wait_asynccnt)
#define USE_ASYNC_LDS 1
#else
#define USE_ASYNC_LDS 0
#endif

typedef __attribute__((address_space(1))) v4i as1_v4i;
typedef __attribute__((address_space(3))) v4i as3_v4i;

__device__ inline void copy16_g2lds(const _Float16* g, _Float16* l)
{
#if USE_ASYNC_LDS
    __builtin_amdgcn_global_load_async_to_lds_b128(
        (as1_v4i*)(uintptr_t)g, (as3_v4i*)(uintptr_t)l, 0, 0);
#else
    *(v8h*)l = *(const v8h*)g;
#endif
}

__device__ inline void async_lds_wait()
{
#if USE_ASYNC_LDS
    __builtin_amdgcn_s_wait_asynccnt(0);
#endif
}

// ---------------------------------------------------------------------------
// Weight packing: f32 row-major (K x 128) -> f16 WMMA B-fragment layout.
// Slot = 4 K-chunks x 8 N-tiles x (32 lanes x 16 halves) = 16384 halves.
// B order used here (contiguous k, same as A): lane l holds column N = l%16;
// half i maps to K = kc*32 + i + 16*(l/16).
// ---------------------------------------------------------------------------
#define SLOT_HALVES 16384

__global__ void k_pack_w(const float* __restrict__ W, _Float16* __restrict__ out,
                         int Kdim, int layers, long slotStrideHalves)
{
    long tid = (long)blockIdx.x * blockDim.x + threadIdx.x;
    long total = (long)layers * SLOT_HALVES;
    if (tid >= total) return;
    int l = (int)(tid / SLOT_HALVES);
    int r = (int)(tid % SLOT_HALVES);
    int kc   = r >> 12;          // 0..3
    int nt   = (r >> 9) & 7;     // 0..7
    int lane = (r >> 4) & 31;    // 0..31
    int i    = r & 15;           // 0..15
    int K = kc * 32 + i + ((lane >> 4) << 4);
    int n = nt * 16 + (lane & 15);
    float v = (K < Kdim) ? W[(long)l * Kdim * HID + (long)K * HID + n] : 0.0f;
    out[(long)l * slotStrideHalves + r] = (_Float16)v;
}

__global__ void k_f32_to_f16(const float* __restrict__ in, _Float16* __restrict__ out, long n)
{
    long t = (long)blockIdx.x * blockDim.x + threadIdx.x;
    if (t < n) out[t] = (_Float16)in[t];
}

__global__ void k_fill(float* __restrict__ p, long n, float v)
{
    long t = (long)blockIdx.x * blockDim.x + threadIdx.x;
    if (t < n) p[t] = v;
}

// ---------------------------------------------------------------------------
// GEMM: C[M x 128] = act( A[M x KDIM](f16, row-major, stride LDA) @ B + bias )
// Block = 256 threads = 8 waves; 128 output rows per block (16 per wave).
// The packed B matrix (KC*8KB) is staged into LDS once per block (async
// global->LDS DMA when available), then all 8 waves read B fragments with
// ds_load_b128. K-loop fully unrolled; A-fragments preloaded from global.
// M must be a multiple of 128 (16384 and 262144 both are).
// ---------------------------------------------------------------------------
template<int KC, int KDIM, int LDA>
__global__ __launch_bounds__(256)
void k_gemm_n128(const _Float16* __restrict__ A,
                 const _Float16* __restrict__ Bp, const float* __restrict__ bias,
                 int relu, float* __restrict__ Cf, _Float16* __restrict__ Ch)
{
    __shared__ _Float16 lB[KC * 8 * 512];    // KC * 8 KB
    const int tid  = threadIdx.x;
    const int lane = threadIdx.x & 31;
    const int wave = threadIdx.x >> 5;
    const long rowBase = (long)blockIdx.x * 128 + (long)wave * 16;
    const int hi16 = lane >> 4;                  // 0 or 1
    const long m = rowBase + (lane & 15);        // A-fragment row for this lane

    // Stage packed B into LDS: KC*2 chunks of 16B per thread.
    #pragma unroll
    for (int c = 0; c < KC * 2; ++c) {
        const int idx = (c * 256 + tid) * 8;     // in halves (16B granules)
        copy16_g2lds(Bp + idx, &lB[idx]);
    }

    // Preload all A fragments (k >= KDIM zero-filled; validity varies with lane half).
    const _Float16* arow = A + m * (long)LDA;
    Frag16 a[KC];
    #pragma unroll
    for (int kc = 0; kc < KC; ++kc) {
        const int k0 = kc * 32 + hi16 * 16;
        if (k0 < KDIM)      a[kc].p.lo = *(const v8h*)(arow + k0);
        else                     a[kc].p.lo = zero8h();
        if (k0 + 8 < KDIM) a[kc].p.hi = *(const v8h*)(arow + k0 + 8);
        else                     a[kc].p.hi = zero8h();
    }

    v8f acc[8];
    #pragma unroll
    for (int t = 0; t < 8; ++t) {
        #pragma unroll
        for (int j = 0; j < 8; ++j) acc[t][j] = 0.0f;
    }

    async_lds_wait();
    __syncthreads();

    #pragma unroll
    for (int kc = 0; kc < KC; ++kc) {
        const _Float16* bchunk = &lB[kc * 4096 + lane * 16];
        Frag16 b[8];
        #pragma unroll
        for (int nt = 0; nt < 8; ++nt) {
            b[nt].p.lo = *(const v8h*)(bchunk + nt * 512);
            b[nt].p.hi = *(const v8h*)(bchunk + nt * 512 + 8);
        }
        #pragma unroll
        for (int nt = 0; nt < 8; ++nt) {
            acc[nt] = __builtin_amdgcn_wmma_f32_16x16x32_f16(
                false, a[kc].v, false, b[nt].v, (short)0, acc[nt], false, false);
        }
    }

    // Epilogue: C/D f32 16x16 layout: lane col N = lane%16; VGPR r -> row M = r + 8*(lane/16)
    #pragma unroll
    for (int nt = 0; nt < 8; ++nt) {
        const int col = nt * 16 + (lane & 15);
        const float bv = bias ? bias[col] : 0.0f;
        #pragma unroll
        for (int r = 0; r < 8; ++r) {
            const long row = rowBase + r + hi16 * 8;
            float v = acc[nt][r] + bv;
            if (relu) v = fmaxf(v, 0.0f);
            if (Cf) Cf[row * HID + col] = v;
            if (Ch) Ch[row * HID + col] = (_Float16)v;
        }
    }
}

// ---------------------------------------------------------------------------
// Attention (segment softmax over incoming edges of each destination node)
// ---------------------------------------------------------------------------
__device__ inline void atomicMaxFloat(float* addr, float val)
{
    // Classic monotone trick: works with addr initialized to -inf.
    if (val >= 0.0f) atomicMax((int*)addr, __float_as_int(val));
    else             atomicMin((unsigned int*)addr, __float_as_uint(val));
}

__global__ void k_attn_alpha(const float* __restrict__ q, const float* __restrict__ k,
                             const float* __restrict__ ep, const int* __restrict__ src,
                             const int* __restrict__ dst, float* __restrict__ alpha,
                             float* __restrict__ amax)
{
    long t = (long)blockIdx.x * blockDim.x + threadIdx.x;
    if (t >= (long)N_EDGES * HEADS) return;
    int e = (int)(t >> 3), hd = (int)(t & 7);
    int s = src[e], d = dst[e];
    const float4* qp  = (const float4*)(q  + (long)d * HID + hd * DH);
    const float4* kp  = (const float4*)(k  + (long)s * HID + hd * DH);
    const float4* epp = (const float4*)(ep + (long)e * HID + hd * DH);
    float dot = 0.0f;
    #pragma unroll
    for (int j = 0; j < 4; ++j) {
        float4 qv = qp[j], kv = kp[j], ev = epp[j];
        dot += qv.x * (kv.x + ev.x) + qv.y * (kv.y + ev.y)
             + qv.z * (kv.z + ev.z) + qv.w * (kv.w + ev.w);
    }
    float al = dot * 0.25f;   // 1/sqrt(DH), DH=16
    alpha[t] = al;
    atomicMaxFloat(&amax[(long)d * HEADS + hd], al);
}

__global__ void k_attn_exp(float* __restrict__ alpha, const float* __restrict__ amax,
                           float* __restrict__ denom, const int* __restrict__ dst)
{
    long t = (long)blockIdx.x * blockDim.x + threadIdx.x;
    if (t >= (long)N_EDGES * HEADS) return;
    int e = (int)(t >> 3), hd = (int)(t & 7);
    int d = dst[e];
    float ex = expf(alpha[t] - amax[(long)d * HEADS + hd]);
    alpha[t] = ex;
    atomicAdd(&denom[(long)d * HEADS + hd], ex);
}

__global__ void k_attn_agg(const float* __restrict__ alpha, const float* __restrict__ denom,
                           const float* __restrict__ v, const float* __restrict__ ep,
                           const int* __restrict__ src, const int* __restrict__ dst,
                           float* __restrict__ out)
{
    long t = (long)blockIdx.x * blockDim.x + threadIdx.x;
    if (t >= (long)N_EDGES * HEADS) return;
    int e = (int)(t >> 3), hd = (int)(t & 7);
    int s = src[e], d = dst[e];
    float a = alpha[t] / (denom[(long)d * HEADS + hd] + 1e-16f);
    const float4* vp  = (const float4*)(v  + (long)s * HID + hd * DH);
    const float4* epp = (const float4*)(ep + (long)e * HID + hd * DH);
    float* op = out + (long)d * HID + hd * DH;
    #pragma unroll
    for (int j = 0; j < 4; ++j) {
        float4 vv = vp[j], ev = epp[j];
        atomicAdd(op + j * 4 + 0, a * (vv.x + ev.x));
        atomicAdd(op + j * 4 + 1, a * (vv.y + ev.y));
        atomicAdd(op + j * 4 + 2, a * (vv.z + ev.z));
        atomicAdd(op + j * 4 + 3, a * (vv.w + ev.w));
    }
}

// ---------------------------------------------------------------------------
// Beta gate: beta = sigmoid([out, xr, out-xr] . wbeta); comb = beta*xr + (1-beta)*out
// One block (128 threads) per node.
// ---------------------------------------------------------------------------
__global__ __launch_bounds__(128)
void k_gate(const float* __restrict__ att, const float* __restrict__ xr,
            const float* __restrict__ wb, _Float16* __restrict__ comb16)
{
    __shared__ float red[128];
    const int n = blockIdx.x, i = threadIdx.x;
    float o = att[(long)n * HID + i];
    float x = xr[(long)n * HID + i];
    red[i] = o * wb[i] + x * wb[HID + i] + (o - x) * wb[2 * HID + i];
    __syncthreads();
    for (int s = 64; s > 0; s >>= 1) {
        if (i < s) red[i] += red[i + s];
        __syncthreads();
    }
    float beta = 1.0f / (1.0f + expf(-red[0]));
    comb16[(long)n * HID + i] = (_Float16)(beta * x + (1.0f - beta) * o);
}

// ---------------------------------------------------------------------------
// Residual + LayerNorm: h = LN(nm + h) * g + b ; writes h (f32) and h16 (f16)
// One block (128 threads) per node.
// ---------------------------------------------------------------------------
__global__ __launch_bounds__(128)
void k_ln(const float* __restrict__ nm, float* __restrict__ h,
          _Float16* __restrict__ h16, const float* __restrict__ g,
          const float* __restrict__ b)
{
    __shared__ float red[128];
    const int n = blockIdx.x, i = threadIdx.x;
    float z = nm[(long)n * HID + i] + h[(long)n * HID + i];
    red[i] = z;
    __syncthreads();
    for (int s = 64; s > 0; s >>= 1) {
        if (i < s) red[i] += red[i + s];
        __syncthreads();
    }
    float mu = red[0] * (1.0f / HID);
    __syncthreads();
    float d = z - mu;
    red[i] = d * d;
    __syncthreads();
    for (int s = 64; s > 0; s >>= 1) {
        if (i < s) red[i] += red[i + s];
        __syncthreads();
    }
    float var = red[0] * (1.0f / HID);
    float y = d * rsqrtf(var + 1e-5f) * g[i] + b[i];
    h[(long)n * HID + i] = y;
    h16[(long)n * HID + i] = (_Float16)y;
}

// ---------------------------------------------------------------------------
// Global mean pool (atomic accumulate), then tiny prediction head.
// ---------------------------------------------------------------------------
__global__ void k_pool(const float* __restrict__ h, const int* __restrict__ batch,
                       float* __restrict__ pool, float* __restrict__ cnt)
{
    long t = (long)blockIdx.x * blockDim.x + threadIdx.x;
    if (t >= (long)N_NODES * 32) return;
    int n = (int)(t >> 5);
    int jb = (int)(t & 31) * 4;
    int b = batch[n];
    const float* hp = h + (long)n * HID + jb;
    float* pp = pool + (long)b * HID + jb;
    atomicAdd(pp + 0, hp[0]);
    atomicAdd(pp + 1, hp[1]);
    atomicAdd(pp + 2, hp[2]);
    atomicAdd(pp + 3, hp[3]);
    if ((t & 31) == 0) atomicAdd(&cnt[b], 1.0f);
}

__global__ void k_head(const float* __restrict__ pool, const float* __restrict__ cnt,
                       const float* __restrict__ w1, const float* __restrict__ b1,
                       const float* __restrict__ w2, const float* __restrict__ b2,
                       float* __restrict__ out)
{
    int g = threadIdx.x;
    if (g >= NGRAPHS) return;
    float inv = 1.0f / fmaxf(cnt[g], 1.0f);
    float o = b2[0];
    for (int m = 0; m < 64; ++m) {
        float t = b1[m];
        for (int j = 0; j < HID; ++j)
            t += pool[(long)g * HID + j] * inv * w1[(long)j * 64 + m];
        t = fmaxf(t, 0.0f);
        o += t * w2[m];
    }
    out[g] = o;
}

// ---------------------------------------------------------------------------
// Launcher
// ---------------------------------------------------------------------------
extern "C" void kernel_launch(void* const* d_in, const int* in_sizes, int n_in,
                              void* d_out, int out_size, void* d_ws, size_t ws_size,
                              hipStream_t stream)
{
    (void)in_sizes; (void)n_in; (void)out_size; (void)ws_size;
    const float* x     = (const float*)d_in[0];
    const float* eattr = (const float*)d_in[1];
    const int*   eidx  = (const int*)d_in[2];
    const int*   batch = (const int*)d_in[3];
    const float* w_ne  = (const float*)d_in[4];  const float* b_ne  = (const float*)d_in[5];
    const float* w_ee  = (const float*)d_in[6];  const float* b_ee  = (const float*)d_in[7];
    const float* w_em1 = (const float*)d_in[8];  const float* b_em1 = (const float*)d_in[9];
    const float* w_em2 = (const float*)d_in[10]; const float* b_em2 = (const float*)d_in[11];
    const float* wq    = (const float*)d_in[12]; const float* bq    = (const float*)d_in[13];
    const float* wk    = (const float*)d_in[14]; const float* bk    = (const float*)d_in[15];
    const float* wv    = (const float*)d_in[16]; const float* bv    = (const float*)d_in[17];
    const float* we    = (const float*)d_in[18];
    const float* wskip = (const float*)d_in[19]; const float* bskip = (const float*)d_in[20];
    const float* wbeta = (const float*)d_in[21];
    const float* w_nm1 = (const float*)d_in[22]; const float* b_nm1 = (const float*)d_in[23];
    const float* w_nm2 = (const float*)d_in[24]; const float* b_nm2 = (const float*)d_in[25];
    const float* ln_g  = (const float*)d_in[26]; const float* ln_b  = (const float*)d_in[27];
    const float* w_p1  = (const float*)d_in[28]; const float* b_p1  = (const float*)d_in[29];
    const float* w_p2  = (const float*)d_in[30]; const float* b_p2  = (const float*)d_in[31];
    const int* src = eidx;
    const int* dst = eidx + N_EDGES;

    // ---- workspace carve-out -------------------------------------------------
    char* ws = (char*)d_ws;
    size_t off = 0;
    auto alloc = [&](size_t bytes) -> char* {
        char* p = ws + off;
        off = (off + bytes + 255) & ~(size_t)255;
        return p;
    };
    _Float16* pw    = (_Float16*)alloc((size_t)38 * SLOT_HALVES * sizeof(_Float16));
    _Float16* x16   = (_Float16*)alloc((size_t)N_NODES * HID * 2);
    _Float16* ea16  = (_Float16*)alloc((size_t)N_EDGES * 16  * 2);
    float*    h     = (float*)   alloc((size_t)N_NODES * HID * 4);
    _Float16* h16   = (_Float16*)alloc((size_t)N_NODES * HID * 2);
    _Float16* e16a  = (_Float16*)alloc((size_t)N_EDGES * HID * 2);
    _Float16* e16b  = (_Float16*)alloc((size_t)N_EDGES * HID * 2);
    float*    ep    = (float*)   alloc((size_t)N_EDGES * HID * 4);
    float*    qb    = (float*)   alloc((size_t)N_NODES * HID * 4);
    float*    kb    = (float*)   alloc((size_t)N_NODES * HID * 4);
    float*    vb    = (float*)   alloc((size_t)N_NODES * HID * 4);
    float*    xr    = (float*)   alloc((size_t)N_NODES * HID * 4);
    float*    att   = (float*)   alloc((size_t)N_NODES * HID * 4);
    float*    alpha = (float*)   alloc((size_t)N_EDGES * HEADS * 4);
    float*    amax  = (float*)   alloc((size_t)N_NODES * HEADS * 4);
    float*    den   = (float*)   alloc((size_t)N_NODES * HEADS * 4);
    _Float16* comb16= (_Float16*)alloc((size_t)N_NODES * HID * 2);
    _Float16* nt16  = (_Float16*)alloc((size_t)N_NODES * HID * 2);
    float*    nm2f  = (float*)   alloc((size_t)N_NODES * HID * 4);
    float*    pool  = (float*)   alloc((size_t)NGRAPHS * HID * 4);
    float*    cnt   = (float*)   alloc((size_t)NGRAPHS * 4);

    auto slot = [&](int s) -> _Float16* { return pw + (size_t)s * SLOT_HALVES; };

    // ---- pack weights into WMMA B-fragment layout ---------------------------
    const dim3 bt(256);
    const dim3 pg1((1 * SLOT_HALVES + 255) / 256);
    const dim3 pg4((NLAYERS * SLOT_HALVES + 255) / 256);
    const long s9 = 9L * SLOT_HALVES;
    k_pack_w<<<pg1, bt, 0, stream>>>(w_ne, slot(0), 128, 1, 0L);
    k_pack_w<<<pg1, bt, 0, stream>>>(w_ee, slot(1), 16, 1, 0L);
    k_pack_w<<<pg4, bt, 0, stream>>>(w_em1, slot(2 + 0), 128, NLAYERS, s9);
    k_pack_w<<<pg4, bt, 0, stream>>>(w_em2, slot(2 + 1), 128, NLAYERS, s9);
    k_pack_w<<<pg4, bt, 0, stream>>>(wq,    slot(2 + 2), 128, NLAYERS, s9);
    k_pack_w<<<pg4, bt, 0, stream>>>(wk,    slot(2 + 3), 128, NLAYERS, s9);
    k_pack_w<<<pg4, bt, 0, stream>>>(wv,    slot(2 + 4), 128, NLAYERS, s9);
    k_pack_w<<<pg4, bt, 0, stream>>>(we,    slot(2 + 5), 128, NLAYERS, s9);
    k_pack_w<<<pg4, bt, 0, stream>>>(wskip, slot(2 + 6), 128, NLAYERS, s9);
    k_pack_w<<<pg4, bt, 0, stream>>>(w_nm1, slot(2 + 7), 128, NLAYERS, s9);
    k_pack_w<<<pg4, bt, 0, stream>>>(w_nm2, slot(2 + 8), 128, NLAYERS, s9);

    // ---- f16 copies of raw inputs -------------------------------------------
    k_f32_to_f16<<<dim3((N_NODES * HID) / 256), bt, 0, stream>>>(x, x16, (long)N_NODES * HID);
    k_f32_to_f16<<<dim3((N_EDGES * 16) / 256), bt, 0, stream>>>(eattr, ea16, (long)N_EDGES * 16);

    // ---- encoders ------------------------------------------------------------
    const dim3 gt(256);
    const dim3 gN(N_NODES / 128);
    const dim3 gE(N_EDGES / 128);
    k_gemm_n128<4, 128, 128><<<gN, gt, 0, stream>>>(x16, slot(0), b_ne, 1, h, h16);
    k_gemm_n128<1, 16, 16><<<gE, gt, 0, stream>>>(ea16, slot(1), b_ee, 0, nullptr, e16a);

    // ---- transformer layers --------------------------------------------------
    const dim3 gEH((N_EDGES * HEADS) / 256);
    for (int l = 0; l < NLAYERS; ++l) {
        const int sb = 2 + l * 9;
        // edge MLP (carried in f16)
        k_gemm_n128<4, 128, 128><<<gE, gt, 0, stream>>>(e16a, slot(sb + 0), b_em1 + l * HID, 1, nullptr, e16b);
        k_gemm_n128<4, 128, 128><<<gE, gt, 0, stream>>>(e16b, slot(sb + 1), b_em2 + l * HID, 0, nullptr, e16a);
        // ep = e @ we (no bias)
        k_gemm_n128<4, 128, 128><<<gE, gt, 0, stream>>>(e16a, slot(sb + 5), nullptr, 0, ep, nullptr);
        // q, k, v, skip projections
        k_gemm_n128<4, 128, 128><<<gN, gt, 0, stream>>>(h16, slot(sb + 2), bq + l * HID, 0, qb, nullptr);
        k_gemm_n128<4, 128, 128><<<gN, gt, 0, stream>>>(h16, slot(sb + 3), bk + l * HID, 0, kb, nullptr);
        k_gemm_n128<4, 128, 128><<<gN, gt, 0, stream>>>(h16, slot(sb + 4), bv + l * HID, 0, vb, nullptr);
        k_gemm_n128<4, 128, 128><<<gN, gt, 0, stream>>>(h16, slot(sb + 6), bskip + l * HID, 0, xr, nullptr);
        // segment softmax attention
        k_fill<<<dim3((N_NODES * HEADS) / 256), bt, 0, stream>>>(amax, (long)N_NODES * HEADS, -INFINITY);
        k_fill<<<dim3((N_NODES * HEADS) / 256), bt, 0, stream>>>(den, (long)N_NODES * HEADS, 0.0f);
        k_fill<<<dim3((N_NODES * HID) / 256), bt, 0, stream>>>(att, (long)N_NODES * HID, 0.0f);
        k_attn_alpha<<<gEH, bt, 0, stream>>>(qb, kb, ep, src, dst, alpha, amax);
        k_attn_exp<<<gEH, bt, 0, stream>>>(alpha, amax, den, dst);
        k_attn_agg<<<gEH, bt, 0, stream>>>(alpha, den, vb, ep, src, dst, att);
        // beta gate
        k_gate<<<dim3(N_NODES), dim3(128), 0, stream>>>(att, xr, wbeta + (long)l * 3 * HID, comb16);
        // node MLP
        k_gemm_n128<4, 128, 128><<<gN, gt, 0, stream>>>(comb16, slot(sb + 7), b_nm1 + l * HID, 1, nullptr, nt16);
        k_gemm_n128<4, 128, 128><<<gN, gt, 0, stream>>>(nt16, slot(sb + 8), b_nm2 + l * HID, 0, nm2f, nullptr);
        // residual + LayerNorm
        k_ln<<<dim3(N_NODES), dim3(128), 0, stream>>>(nm2f, h, h16, ln_g + l * HID, ln_b + l * HID);
    }

    // ---- mean pool + head ----------------------------------------------------
    k_fill<<<dim3((NGRAPHS * HID) / 256 + 1), bt, 0, stream>>>(pool, (long)NGRAPHS * HID, 0.0f);
    k_fill<<<dim3(1), bt, 0, stream>>>(cnt, (long)NGRAPHS, 0.0f);
    k_pool<<<dim3((N_NODES * 32) / 256), bt, 0, stream>>>(h, batch, pool, cnt);
    k_head<<<dim3(1), dim3(64), 0, stream>>>(pool, cnt, w_p1, b_p1, w_p2, b_p2, (float*)d_out);
}
